// QuantumAttentionWithGroverAndQAOA_13709535609630
// MI455X (gfx1250) — matrix operand order repaired and checked by reading, not compiled, not run on hardware
//
#include <hip/hip_runtime.h>
#include <math.h>

typedef __attribute__((ext_vector_type(16))) _Float16 v16h;
typedef __attribute__((ext_vector_type(8)))  float    v8f;

#define NQb 4
#define NLl 3
#define Bn  8
#define Tn  512
#define Fn  64
#define Sn  16
#define Dn  192            // S * 12
#define Nn  4096           // B * T
#define Un  48             // 3 * S
#define PI_F 3.14159265358979f

struct c32 { float x, y; };

// ---------------- quantum gate helpers (state = 16 complex amps in regs) ----
// wire q <-> bit (3-q) of the flat index (state reshaped (2,2,2,2), wire 0 = MSB)

__device__ __forceinline__ void g_rx(c32* st, int q, float t) {
  float h = 0.5f * t, c = __cosf(h), s = __sinf(h);
  int m = 8 >> q;
#pragma unroll
  for (int k = 0; k < 16; ++k) {
    if (k & m) continue;
    c32 a = st[k], b = st[k | m];
    st[k]     = { c * a.x + s * b.y, c * a.y - s * b.x };  // c*a + (-i s)*b
    st[k | m] = { c * b.x + s * a.y, c * b.y - s * a.x };
  }
}

__device__ __forceinline__ void g_ry(c32* st, int q, float t) {
  float h = 0.5f * t, c = __cosf(h), s = __sinf(h);
  int m = 8 >> q;
#pragma unroll
  for (int k = 0; k < 16; ++k) {
    if (k & m) continue;
    c32 a = st[k], b = st[k | m];
    st[k]     = { c * a.x - s * b.x, c * a.y - s * b.y };
    st[k | m] = { s * a.x + c * b.x, s * a.y + c * b.y };
  }
}

__device__ __forceinline__ void g_rz(c32* st, int q, float t) {
  float h = 0.5f * t, c = __cosf(h), s = __sinf(h);
  int m = 8 >> q;
#pragma unroll
  for (int k = 0; k < 16; ++k) {
    c32 a = st[k];
    if (k & m) st[k] = { c * a.x - s * a.y, c * a.y + s * a.x };  // e^{+i h}
    else       st[k] = { c * a.x + s * a.y, c * a.y - s * a.x };  // e^{-i h}
  }
}

__device__ __forceinline__ void g_h(c32* st, int q) {
  const float r = 0.70710678118654752f;
  int m = 8 >> q;
#pragma unroll
  for (int k = 0; k < 16; ++k) {
    if (k & m) continue;
    c32 a = st[k], b = st[k | m];
    st[k]     = { (a.x + b.x) * r, (a.y + b.y) * r };
    st[k | m] = { (a.x - b.x) * r, (a.y - b.y) * r };
  }
}

__device__ __forceinline__ void g_x(c32* st, int q) {
  int m = 8 >> q;
#pragma unroll
  for (int k = 0; k < 16; ++k) {
    if (k & m) continue;
    c32 a = st[k]; st[k] = st[k | m]; st[k | m] = a;
  }
}

__device__ __forceinline__ void g_crx(c32* st, int qc, int qt, float t) {
  float h = 0.5f * t, c = __cosf(h), s = __sinf(h);
  int mc = 8 >> qc, mt = 8 >> qt;
#pragma unroll
  for (int k = 0; k < 16; ++k) {
    if (!(k & mc) || (k & mt)) continue;
    c32 a = st[k], b = st[k | mt];
    st[k]      = { c * a.x + s * b.y, c * a.y - s * b.x };
    st[k | mt] = { c * b.x + s * a.y, c * b.y - s * a.x };
  }
}

__device__ __forceinline__ void g_cnot(c32* st, int qc, int qt) {
  int mc = 8 >> qc, mt = 8 >> qt;
#pragma unroll
  for (int k = 0; k < 16; ++k) {
    if (!(k & mc) || (k & mt)) continue;
    c32 a = st[k]; st[k] = st[k | mt]; st[k | mt] = a;
  }
}

__device__ __forceinline__ void g_iswap(c32* st, int q0, int q1) {
  int m0 = 8 >> q0, m1 = 8 >> q1;
#pragma unroll
  for (int k = 0; k < 16; ++k) {
    if ((k & m0) || (k & m1)) continue;
    c32 a01 = st[k | m1], a10 = st[k | m0];
    st[k | m1] = { -a10.y, a10.x };   // i * a10
    st[k | m0] = { -a01.y, a01.x };   // i * a01
  }
}

__device__ __forceinline__ void g_cphase(c32* st, int q0, int q1, float phi) {
  float c = __cosf(phi), s = __sinf(phi);
  int m0 = 8 >> q0, m1 = 8 >> q1;
#pragma unroll
  for (int k = 0; k < 16; ++k) {
    if (!(k & m0) || !(k & m1)) continue;
    c32 a = st[k];
    st[k] = { c * a.x - s * a.y, c * a.y + s * a.x };
  }
}

__device__ __forceinline__ void g_swap(c32* st, int q0, int q1) {
  int m0 = 8 >> q0, m1 = 8 >> q1;
#pragma unroll
  for (int k = 0; k < 16; ++k) {
    if ((k & m0) || (k & m1)) continue;
    c32 a = st[k | m0]; st[k | m0] = st[k | m1]; st[k | m1] = a;
  }
}

// ---------------- kernel 1: circuit simulation, one thread per (u, n) -------
// q, k written row-major per (g, n, d); v written TRANSPOSED per batch: vT[b][d][t]
__global__ __launch_bounds__(256) void qcirc_kernel(
    const float* __restrict__ x,      // (4096, 64)
    const float* __restrict__ w,      // (48, 3, 4)
    const float* __restrict__ ent,    // (48, 3, 4)
    const float* __restrict__ tun,    // (48, 3, 4)
    const float* __restrict__ betas,  // (48, 3)
    const float* __restrict__ gammas, // (48, 3)
    _Float16* __restrict__ qkvh)      // q,k: (2, 4096, 192); vT: (8, 192, 512)
{
  int idx = blockIdx.x * blockDim.x + threadIdx.x;
  if (idx >= Un * Nn) return;
  int u = idx / Nn;
  int n = idx - u * Nn;
  int g = u / Sn, s = u - g * Sn;

  float xa[4];
#pragma unroll
  for (int q = 0; q < 4; ++q) xa[q] = x[n * Fn + s * 4 + q];

  float wL[NLl][4], eL[NLl][4], tL[NLl][4], bL[NLl], gL[NLl];
#pragma unroll
  for (int l = 0; l < NLl; ++l) {
#pragma unroll
    for (int q = 0; q < 4; ++q) {
      wL[l][q] = w[(u * NLl + l) * 4 + q];
      eL[l][q] = ent[(u * NLl + l) * 4 + q];
      tL[l][q] = tun[(u * NLl + l) * 4 + q];
    }
    bL[l] = betas[u * NLl + l];
    gL[l] = gammas[u * NLl + l];
  }

  c32 st[16];
#pragma unroll
  for (int k = 0; k < 16; ++k) st[k] = { 0.f, 0.f };
  st[0].x = 1.f;

#pragma unroll
  for (int q = 0; q < 4; ++q) g_rx(st, q, xa[q]);

#pragma unroll
  for (int l = 0; l < NLl; ++l) {
    // parametrized single-qubit rotations
#pragma unroll
    for (int q = 0; q < 4; ++q) {
      float th = wL[l][q];
      g_rx(st, q, th); g_ry(st, q, th); g_rz(st, q, th);
    }
    // entangling CRX ring (q -> q+2, q -> q+3)
#pragma unroll
    for (int q = 0; q < 4; ++q) {
      float e = eL[l][q];
      g_crx(st, q, (q + 2) & 3, e);
      g_crx(st, q, (q + 3) & 3, e);
    }
    // iswap chain
#pragma unroll
    for (int q = 0; q < 3; ++q) g_iswap(st, q, q + 1);
    g_iswap(st, 3, 0);
    // tunneling CRX ring
#pragma unroll
    for (int q = 0; q < 3; ++q) g_crx(st, q, q + 1, tL[l][q]);
    g_crx(st, 3, 0, tL[l][3]);
    // Grover diffusion-like block: H X ... mcx ... X H
#pragma unroll
    for (int q = 0; q < 4; ++q) { g_h(st, q); g_x(st, q); }
    { c32 a = st[14]; st[14] = st[15]; st[15] = a; }  // mcx (c=q0,q1,q2 t=q3)
#pragma unroll
    for (int q = 0; q < 4; ++q) { g_x(st, q); g_h(st, q); }
#pragma unroll
    for (int q = 0; q < 4; ++q) g_h(st, q);
    // QAOA-like block
    {
      float bb = bL[l], gg = gL[l];
#pragma unroll
      for (int q = 0; q < 4; ++q) {
        g_cnot(st, q, (q + 1) & 3);
        g_rz(st, q, 2.f * bb * wL[l][q]);
      }
#pragma unroll
      for (int q = 0; q < 4; ++q) g_rx(st, q, gg);
    }
#pragma unroll
    for (int q = 0; q < 4; ++q) g_h(st, q);
    // QFT-like phases
#pragma unroll
    for (int i = 0; i < 4; ++i)
#pragma unroll
      for (int j = 0; j < 4; ++j)
        if (j < i) g_cphase(st, j, i, PI_F / (float)(1 << (i - j)));
    g_swap(st, 0, 3); g_swap(st, 1, 2);
#pragma unroll
    for (int i = 0; i < 4; ++i) {
      g_h(st, i);
#pragma unroll
      for (int j = 0; j < 4; ++j)
        if (j < i) g_cphase(st, j, i, -PI_F / (float)(1 << (i - j)));
    }
  }

  // measurements: Z_i, <flip_i> (X-like), <i-flip_i> (Y-like)
  float outv[12];
#pragma unroll
  for (int i = 0; i < 4; ++i) {
    int m = 8 >> i;
    float z = 0.f, xe = 0.f, ye = 0.f;
#pragma unroll
    for (int k = 0; k < 16; ++k) {
      c32 a = st[k];
      float p = a.x * a.x + a.y * a.y;
      z += (k & m) ? -p : p;
      c32 bv = st[k ^ m];
      xe += a.x * bv.x + a.y * bv.y;
      float t = a.x * bv.y - a.y * bv.x;
      ye += (k & m) ? -t : t;
    }
    outv[i] = z; outv[4 + i] = xe; outv[8 + i] = ye;
  }

  if (g < 2) {
    // q, k: row-major [g][n][d]
    _Float16* op = qkvh + (size_t)g * (Nn * Dn) + (size_t)n * Dn + s * 12;
#pragma unroll
    for (int o = 0; o < 12; ++o) op[o] = (_Float16)outv[o];
  } else {
    // v: transposed per batch vT[b][d][t]  (so attn@v B-fragments are contiguous)
    int bb = n >> 9, tt = n & 511;
    _Float16* vp = qkvh + (size_t)2 * (Nn * Dn)
                 + ((size_t)bb * Dn + s * 12) * Tn + tt;
#pragma unroll
    for (int o = 0; o < 12; ++o) vp[(size_t)o * Tn] = (_Float16)outv[o];
  }
}

// ---------------- WMMA fragment loaders (CDNA5 wave32 layouts) --------------
// A 16x32 f16: lane m (0-15) holds row m; half_id selects K group 0-15 vs 16-31
__device__ __forceinline__ v16h load_a_frag(const _Float16* rowp, int half_id) {
  v16h a;
  _Float16* ap = (_Float16*)&a;
#pragma unroll
  for (int v = 0; v < 8; ++v) {
    int kb = half_id * 16 + v * 2;
    ap[2 * v]     = rowp[kb];
    ap[2 * v + 1] = rowp[kb + 1];
  }
  return a;
}

// B 32x16 f16 from a matrix laid out with K contiguous per output column:
// lane holds column N = lane%16; VGPR v holds K = half_id*16 + 2v, 2v+1
__device__ __forceinline__ v16h load_b_frag_kcontig(const _Float16* rowp) {
  v16h bf;
  _Float16* bp = (_Float16*)&bf;
#pragma unroll
  for (int v = 0; v < 16; ++v) bp[v] = rowp[v];
  return bf;
}

// ---------------- kernel 2: attention + layernorm, one wave per 16-row tile -
__global__ __launch_bounds__(32) void attn_kernel(
    const _Float16* __restrict__ qkvh,    // q,k row-major; vT transposed
    const float* __restrict__ ln_gamma,   // (192,)
    const float* __restrict__ ln_beta,    // (192,)
    float* __restrict__ y_out,            // (8, 512, 192)
    float* __restrict__ attn_out)         // (8, 512, 512)
{
  __shared__ float    sS[16 * 512];       // scores f32, later ao f32 (16x192)
  __shared__ _Float16 sA[16 * 512];       // attn probs f16

  const int lane    = threadIdx.x;
  const int half_id = lane >> 4;
  const int mrow    = lane & 15;
  const int b       = blockIdx.x >> 5;    // batch
  const int row0    = (blockIdx.x & 31) * 16;

  const _Float16* qh = qkvh + 0 * (size_t)(Nn * Dn) + (size_t)b * Tn * Dn;
  const _Float16* kh = qkvh + 1 * (size_t)(Nn * Dn) + (size_t)b * Tn * Dn;
  const _Float16* vt = qkvh + 2 * (size_t)(Nn * Dn) + (size_t)b * Dn * Tn; // vT[d][t]

  // preload A fragments of q rows [row0, row0+16), full K=192 (6 frags)
  v16h aq[6];
#pragma unroll
  for (int f = 0; f < 6; ++f)
    aq[f] = load_a_frag(qh + (row0 + mrow) * Dn + f * 32, half_id);

  // ---- scores = q @ k^T / sqrt(d) ----
  for (int ct = 0; ct < 32; ++ct) {
    if (ct < 31) // prefetch next tile's k rows (global_prefetch_b8)
      __builtin_prefetch(kh + ((ct + 1) * 16 + mrow) * Dn, 0, 1);
    v8f acc = { 0.f, 0.f, 0.f, 0.f, 0.f, 0.f, 0.f, 0.f };
#pragma unroll
    for (int f = 0; f < 6; ++f) {
      // B[K][N] = k[ct*16+N][f*32+K]: K contiguous along k's row
      v16h bf = load_b_frag_kcontig(
          kh + (ct * 16 + mrow) * Dn + f * 32 + half_id * 16);
      acc = __builtin_amdgcn_wmma_f32_16x16x32_f16(false, aq[f], false, bf,
                                                   (short)0, acc, false, false);
    }
#pragma unroll
    for (int r = 0; r < 8; ++r) {
      int m = r + 8 * half_id;            // C layout: M = r + 8*(lane/16), N = lane%16
      sS[m * 512 + ct * 16 + mrow] = acc[r] * 0.5f;   // 1/SQRT_D
    }
  }
  __syncthreads();

  // ---- softmax: each row split across the two lane halves, shfl_xor(16) ----
  {
    float* row = sS + mrow * 512;
    const int base = half_id * 256;
    float mx = -1e30f;
    for (int t = 0; t < 256; ++t) mx = fmaxf(mx, row[base + t]);
    mx = fmaxf(mx, __shfl_xor(mx, 16));
    float sum = 0.f;
    for (int t = 0; t < 256; ++t) {
      float e = __expf(row[base + t] - mx);
      row[base + t] = e;
      sum += e;
    }
    sum += __shfl_xor(sum, 16);
    float inv = 1.f / sum;
    float* aop = attn_out + ((size_t)b * Tn + row0 + mrow) * Tn + base;
    _Float16* ah = sA + mrow * 512 + base;
    for (int t = 0; t < 256; ++t) {
      float p = row[base + t] * inv;
      aop[t] = p;
      ah[t] = (_Float16)p;
    }
  }
  __syncthreads();

  // ---- ao = attn @ v  (K = 512), 2 passes x 6 column tiles of 16 ----
#pragma unroll
  for (int hp = 0; hp < 2; ++hp) {
    v8f acc[6];
#pragma unroll
    for (int j = 0; j < 6; ++j) acc[j] = (v8f){ 0.f, 0.f, 0.f, 0.f, 0.f, 0.f, 0.f, 0.f };
    for (int kk = 0; kk < 16; ++kk) {   // 512 / 32
      v16h af = load_a_frag(sA + mrow * 512 + kk * 32, half_id);
#pragma unroll
      for (int j = 0; j < 6; ++j) {
        int n0 = (hp * 6 + j) * 16;
        // B[K=t][N=d] = vT[d][t]: K contiguous along vT's row
        v16h bf = load_b_frag_kcontig(
            vt + (size_t)(n0 + mrow) * Tn + kk * 32 + half_id * 16);
        acc[j] = __builtin_amdgcn_wmma_f32_16x16x32_f16(false, af, false, bf,
                                                        (short)0, acc[j], false, false);
      }
    }
#pragma unroll
    for (int j = 0; j < 6; ++j) {
      int n0 = (hp * 6 + j) * 16;
#pragma unroll
      for (int r = 0; r < 8; ++r) {
        int m = r + 8 * half_id;
        sS[m * Dn + n0 + mrow] = acc[j][r];   // reuse sS as ao f32 (16 x 192)
      }
    }
  }
  __syncthreads();

  // ---- LayerNorm over D=192: split row across lane halves, shfl_xor(16) ----
  {
    const float* ao = sS + mrow * Dn;
    const int base = half_id * 96;
    float mu = 0.f;
    for (int d = 0; d < 96; ++d) mu += ao[base + d];
    mu += __shfl_xor(mu, 16);
    mu *= (1.f / Dn);
    float var = 0.f;
    for (int d = 0; d < 96; ++d) { float t = ao[base + d] - mu; var += t * t; }
    var += __shfl_xor(var, 16);
    var *= (1.f / Dn);
    float rstd = rsqrtf(var + 1e-5f);
    float* yo = y_out + ((size_t)b * Tn + row0 + mrow) * Dn + base;
    for (int d = 0; d < 96; ++d)
      yo[d] = (ao[base + d] - mu) * rstd * ln_gamma[base + d] + ln_beta[base + d];
  }
}

// ---------------- launch ----------------------------------------------------
extern "C" void kernel_launch(void* const* d_in, const int* in_sizes, int n_in,
                              void* d_out, int out_size, void* d_ws, size_t ws_size,
                              hipStream_t stream) {
  const float* x        = (const float*)d_in[0];
  const float* w        = (const float*)d_in[1];
  const float* ent      = (const float*)d_in[2];
  const float* tun      = (const float*)d_in[3];
  const float* betas    = (const float*)d_in[4];
  const float* gammas   = (const float*)d_in[5];
  const float* ln_gamma = (const float*)d_in[6];
  const float* ln_beta  = (const float*)d_in[7];

  _Float16* qkvh = (_Float16*)d_ws;                       // 3*4096*192 halves
  float* y_out    = (float*)d_out;                        // 8*512*192
  float* attn_out = y_out + (size_t)Bn * Tn * Dn;         // 8*512*512

  int total = Un * Nn;
  qcirc_kernel<<<(total + 255) / 256, 256, 0, stream>>>(
      x, w, ent, tun, betas, gammas, qkvh);
  attn_kernel<<<Bn * 32, 32, 0, stream>>>(
      qkvh, ln_gamma, ln_beta, y_out, attn_out);
}
